// Weird_Model_4690104287849
// MI455X (gfx1250) — matrix operand order repaired and checked by reading, not compiled, not run on hardware
//
#include <hip/hip_runtime.h>
#include <math.h>

// ---------------------------------------------------------------------------
// Types for WMMA fragments (CDNA5 / gfx1250, wave32)
// ---------------------------------------------------------------------------
typedef _Float16 half8_t  __attribute__((ext_vector_type(8)));
typedef _Float16 half16_t __attribute__((ext_vector_type(16)));
typedef float    float8_t __attribute__((ext_vector_type(8)));

__device__ __forceinline__ float gelu_tanh_f(float x) {
    float x3 = x * x * x;
    return 0.5f * x * (1.0f + tanhf(0.7978845608028654f * (x + 0.044715f * x3)));
}

// LDS byte offset of a generic pointer to __shared__ (ISA: LDS aperture keeps
// the LDS offset in the low 32 address bits).
__device__ __forceinline__ unsigned lds_off(const void* p) {
    return (unsigned)(unsigned long long)p;
}

// Async global->LDS 16B copy (gfx1250; tracked by ASYNCcnt).
__device__ __forceinline__ void async_copy_b128(unsigned lds_addr, const void* gptr) {
    unsigned long long ga = (unsigned long long)gptr;
    asm volatile("global_load_async_to_lds_b128 %0, %1, off"
                 :: "v"(lds_addr), "v"(ga) : "memory");
}
__device__ __forceinline__ void wait_asynccnt0() {
    asm volatile("s_wait_asynccnt 0" ::: "memory");
}

// ---------------------------------------------------------------------------
// Generic batched WMMA GEMM:  C = epilogue( scale * A @ B )
//   A: [M,K] f16 (row stride lda)
//   B: [K,N] f16 (row stride ldb)  or, if BTRANS, [N,K] f16 -> computes A@B^T
//   C: [M,N] f16 (or f32 if OUTF32), row stride ldc
// Batch via blockIdx.z -> (b,h) pointer offsets (heads = 64-col slices).
// 256 threads = 8 waves (4 wave-rows x 2 wave-cols); block tile BM x BN,
// BK=32 (one v_wmma_f32_16x16x32_f16 K-step), double-buffered LDS.
// A (and B when BTRANS) staged with async global->LDS b128; NN-B staged via
// registers issued before the WMMA block and committed to LDS after it.
// FLAGS: 1=bias, 2=gelu, 4=accumulate into existing f16 C, 8=store f32,
//        16=B is [N,K]
// ---------------------------------------------------------------------------
template <int FLAGS, int BM, int BN>
__global__ void __launch_bounds__(256)
gemm_wmma_kernel(const _Float16* __restrict__ A,
                 const _Float16* __restrict__ Bm,
                 const float* __restrict__ bias,
                 void* __restrict__ Cv,
                 int M, int N, int K,
                 int lda, int ldb, int ldc,
                 long offA_b, long offA_h,
                 long offB_b, long offB_h,
                 long offC_b, long offC_h,
                 int nheads, float scale)
{
    constexpr bool HAS_BIAS = (FLAGS & 1)  != 0;
    constexpr bool DO_GELU  = (FLAGS & 2)  != 0;
    constexpr bool ACCUM    = (FLAGS & 4)  != 0;
    constexpr bool OUTF32   = (FLAGS & 8)  != 0;
    constexpr bool BTRANS   = (FLAGS & 16) != 0;

    constexpr int LROW = 40;           // padded LDS row (halfs): 80B, 16B-aligned
    constexpr int WM = BM / 4;         // 32 rows per wave
    constexpr int WN = BN / 2;         // 64 (or 32) cols per wave
    constexpr int MI = WM / 16;        // 2
    constexpr int NI = WN / 16;        // 4 (or 2)
    constexpr int CA = BM / 64;        // async b128 chunks per thread for A
    constexpr int CB = BN / 64;        // chunks per thread for B (BT) / half8 regs (NN)

    __shared__ _Float16 As[2 * BM * LROW];
    __shared__ _Float16 Bs[2 * BN * LROW];

    const int tid  = threadIdx.x;
    const int lane = tid & 31;
    const int wave = tid >> 5;
    const int wr   = wave & 3;         // 0..3
    const int wc   = wave >> 2;        // 0..1
    const int r16  = lane & 15;
    const int hf   = lane >> 4;

    const int n0 = blockIdx.x * BN;
    const int m0 = blockIdx.y * BM;

    const int z  = blockIdx.z;
    const int bb = z / nheads;
    const int hh = z - bb * nheads;
    A  += (size_t)bb * offA_b + (size_t)hh * offA_h;
    Bm += (size_t)bb * offB_b + (size_t)hh * offB_h;
    const size_t coff = (size_t)bb * offC_b + (size_t)hh * offC_h;

    float8_t acc[MI][NI];
#pragma unroll
    for (int i = 0; i < MI; ++i)
#pragma unroll
        for (int j = 0; j < NI; ++j)
            acc[i][j] = (float8_t){0.f, 0.f, 0.f, 0.f, 0.f, 0.f, 0.f, 0.f};

    // ---- staging helpers -------------------------------------------------
    auto stageA_async = [&](int buf, int k0) {
#pragma unroll
        for (int c = 0; c < CA; ++c) {
            int idx = tid + c * 256;                 // chunk of 8 halfs
            int m   = idx >> 2;
            int kc  = (idx & 3) * 8;
            async_copy_b128(lds_off(&As[(buf * BM + m) * LROW + kc]),
                            A + (size_t)(m0 + m) * lda + (k0 + kc));
        }
    };
    auto stageBT_async = [&](int buf, int k0) {
#pragma unroll
        for (int c = 0; c < CB; ++c) {
            int idx = tid + c * 256;
            int n   = idx >> 2;
            int kc  = (idx & 3) * 8;
            async_copy_b128(lds_off(&Bs[(buf * BN + n) * LROW + kc]),
                            Bm + (size_t)(n0 + n) * ldb + (k0 + kc));
        }
    };

    half8_t regB[CB];                                // NN-B prefetch registers
    auto issueBN = [&](int k0) {                     // coalesced global loads
#pragma unroll
        for (int c = 0; c < CB; ++c) {
            int idx = tid + c * 256;
            int k   = idx / (BN / 8);
            int nc  = (idx % (BN / 8)) * 8;
            regB[c] = *(const half8_t*)(Bm + (size_t)(k0 + k) * ldb + (n0 + nc));
        }
    };
    auto commitBN = [&](int buf) {                   // LDS transpose commit
#pragma unroll
        for (int c = 0; c < CB; ++c) {
            int idx = tid + c * 256;
            int k   = idx / (BN / 8);
            int nc  = (idx % (BN / 8)) * 8;
#pragma unroll
            for (int j = 0; j < 8; ++j)
                Bs[(buf * BN + nc + j) * LROW + k] = regB[c][j];
        }
    };

    // ---- prologue: stage k=0 into buffer 0 -------------------------------
    stageA_async(0, 0);
    if (BTRANS) {
        stageBT_async(0, 0);
    } else {
        issueBN(0);
        commitBN(0);
    }

    const int nk = K / 32;
    for (int kt = 0; kt < nk; ++kt) {
        const int cur = kt & 1;
        const int nxt = cur ^ 1;
        wait_asynccnt0();
        __syncthreads();                             // cur buffers ready for all

        const bool pf = (kt + 1) < nk;
        if (pf) {                                    // prefetch next tile
            stageA_async(nxt, (kt + 1) * 32);
            if (BTRANS) stageBT_async(nxt, (kt + 1) * 32);
            else        issueBN((kt + 1) * 32);
        }

        // ---- fragments + MIxNI WMMA on cur -------------------------------
        half16_t afr[MI], bfr[NI];
#pragma unroll
        for (int mi = 0; mi < MI; ++mi) {
            // A 16x32 layout: lane-half hf holds K = hf*8..+7 , 16+hf*8..+7
            const _Float16* ap = &As[(cur * BM + wr * WM + mi * 16 + r16) * LROW];
            half8_t alo = *(const half8_t*)(ap + hf * 8);
            half8_t ahi = *(const half8_t*)(ap + 16 + hf * 8);
            afr[mi] = __builtin_shufflevector(alo, ahi,
                        0, 1, 2, 3, 4, 5, 6, 7, 8, 9, 10, 11, 12, 13, 14, 15);
        }
#pragma unroll
        for (int ni = 0; ni < NI; ++ni) {
            // B 32x16 layout: lane-half hf holds K = hf*8..+7 , 16+hf*8..+7
            const _Float16* bp = &Bs[(cur * BN + wc * WN + ni * 16 + r16) * LROW + hf * 8];
            half8_t blo = *(const half8_t*)(bp);
            half8_t bhi = *(const half8_t*)(bp + 16);
            bfr[ni] = __builtin_shufflevector(blo, bhi,
                        0, 1, 2, 3, 4, 5, 6, 7, 8, 9, 10, 11, 12, 13, 14, 15);
        }
#pragma unroll
        for (int mi = 0; mi < MI; ++mi)
#pragma unroll
            for (int ni = 0; ni < NI; ++ni)
                acc[mi][ni] = __builtin_amdgcn_wmma_f32_16x16x32_f16(
                    false, afr[mi], false, bfr[ni], (short)0, acc[mi][ni],
                    false, false);

        if (pf && !BTRANS) commitBN(nxt);            // LDS commit after the math
        __syncthreads();
    }

    // ---- epilogue: D layout — VGPR r holds row r + 8*hf, col = lane&15 ----
#pragma unroll
    for (int mi = 0; mi < MI; ++mi) {
#pragma unroll
        for (int ni = 0; ni < NI; ++ni) {
            int colg = n0 + wc * WN + ni * 16 + r16;
#pragma unroll
            for (int r = 0; r < 8; ++r) {
                int rowg = m0 + wr * WM + mi * 16 + hf * 8 + r;
                float v = acc[mi][ni][r] * scale;
                if (HAS_BIAS) v += bias[colg];
                if (DO_GELU)  v = gelu_tanh_f(v);
                size_t idx = coff + (size_t)rowg * ldc + colg;
                if (ACCUM)  v += (float)((const _Float16*)Cv)[idx];
                if (OUTF32) ((float*)Cv)[idx] = v;
                else        ((_Float16*)Cv)[idx] = (_Float16)v;
            }
        }
    }
}

// ---------------------------------------------------------------------------
// Row softmax, in place, f16 storage / f32 math. One block (256 thr) per row;
// vectorized b128 access (ncols must be exactly 2048 here: ncols=S).
// ---------------------------------------------------------------------------
__global__ void __launch_bounds__(256)
softmax_rows_kernel(_Float16* __restrict__ P, int ncols)
{
    __shared__ float red[256];
    const size_t base = (size_t)blockIdx.x * (size_t)ncols;
    const int tid = threadIdx.x;

    half8_t v8 = *(const half8_t*)(P + base + tid * 8);
    float x[8];
#pragma unroll
    for (int j = 0; j < 8; ++j) x[j] = (float)v8[j];

    float lmax = x[0];
#pragma unroll
    for (int j = 1; j < 8; ++j) lmax = fmaxf(lmax, x[j]);
    red[tid] = lmax;
    __syncthreads();
    for (int s = 128; s > 0; s >>= 1) {
        if (tid < s) red[tid] = fmaxf(red[tid], red[tid + s]);
        __syncthreads();
    }
    const float m = red[0];
    __syncthreads();

    float lsum = 0.f;
#pragma unroll
    for (int j = 0; j < 8; ++j) { x[j] = __expf(x[j] - m); lsum += x[j]; }
    red[tid] = lsum;
    __syncthreads();
    for (int s = 128; s > 0; s >>= 1) {
        if (tid < s) red[tid] += red[tid + s];
        __syncthreads();
    }
    const float inv = 1.0f / red[0];

    half8_t o;
#pragma unroll
    for (int j = 0; j < 8; ++j) o[j] = (_Float16)(x[j] * inv);
    *(half8_t*)(P + base + tid * 8) = o;
}

// ---------------------------------------------------------------------------
// f32 -> f16 conversion (grid-stride)
// ---------------------------------------------------------------------------
__global__ void __launch_bounds__(256)
f32_to_f16_kernel(const float* __restrict__ src, _Float16* __restrict__ dst,
                  size_t n)
{
    size_t i = (size_t)blockIdx.x * blockDim.x + threadIdx.x;
    size_t stride = (size_t)gridDim.x * blockDim.x;
    for (; i < n; i += stride) dst[i] = (_Float16)src[i];
}

// ---------------------------------------------------------------------------
// Host side
// ---------------------------------------------------------------------------
template <int FLAGS, int BM, int BN>
static void launch_gemm(const _Float16* A, const _Float16* B, const float* bias,
                        void* C, int M, int N, int K,
                        int lda, int ldb, int ldc,
                        long aB, long aH, long bB, long bH, long cB, long cH,
                        int nheads, float scale, int batches, hipStream_t s)
{
    dim3 grid(N / BN, M / BM, batches);
    gemm_wmma_kernel<FLAGS, BM, BN><<<grid, 256, 0, s>>>(
        A, B, bias, C, M, N, K, lda, ldb, ldc,
        aB, aH, bB, bH, cB, cH, nheads, scale);
}

extern "C" void kernel_launch(void* const* d_in, const int* in_sizes, int n_in,
                              void* d_out, int out_size, void* d_ws, size_t ws_size,
                              hipStream_t stream)
{
    (void)in_sizes; (void)n_in; (void)out_size; (void)ws_size;

    constexpr int B_  = 8, S = 2048, D = 256, H = 4, DH = 64, DFF = 1024;
    const size_t BSD  = (size_t)B_ * S * D;          // 4,194,304
    const size_t SS   = (size_t)S * S;
    const size_t BHSS = (size_t)B_ * H * SS;         // 134,217,728
    const size_t DD   = (size_t)D * D;

    const float* gf  = (const float*)d_in[0];
    const float* lf  = (const float*)d_in[1];
    const float* tf  = (const float*)d_in[2];
    const float* Wl1 = (const float*)d_in[3];  const float* bl1 = (const float*)d_in[4];
    const float* Wl2 = (const float*)d_in[5];  const float* bl2 = (const float*)d_in[6];
    const float* Wg  = (const float*)d_in[7];  const float* bg  = (const float*)d_in[8];
    const float* Wgq = (const float*)d_in[9];  const float* bgq = (const float*)d_in[10];
    const float* Wgk = (const float*)d_in[11]; const float* bgk = (const float*)d_in[12];
    const float* Wgv = (const float*)d_in[13]; const float* bgv = (const float*)d_in[14];
    const float* Wpq = (const float*)d_in[15]; const float* bpq = (const float*)d_in[16];
    const float* Wpk = (const float*)d_in[17]; const float* bpk = (const float*)d_in[18];
    const float* Wpv = (const float*)d_in[19]; const float* bpv = (const float*)d_in[20];
    const float* Wd  = (const float*)d_in[21]; const float* bd  = (const float*)d_in[22];
    const float* Wm1 = (const float*)d_in[23]; const float* bm1 = (const float*)d_in[24];
    const float* Wm2 = (const float*)d_in[25]; const float* bm2 = (const float*)d_in[26];
    const float* Wml = (const float*)d_in[27]; const float* bml = (const float*)d_in[28];

    // ---- workspace carve (f16 elements, 32B aligned slots) ----
    _Float16* w = (_Float16*)d_ws;
    size_t off = 0;
    auto alloc = [&](size_t n) {
        _Float16* p = w + off;
        off += (n + 15) & ~((size_t)15);
        return p;
    };
    _Float16* hgf  = alloc(BSD);
    _Float16* hlf  = alloc(BSD);
    _Float16* htf  = alloc(BSD);
    _Float16* hWl1 = alloc(DD); _Float16* hWl2 = alloc(DD);
    _Float16* hWg  = alloc(DD);
    _Float16* hWgq = alloc(DD); _Float16* hWgk = alloc(DD); _Float16* hWgv = alloc(DD);
    _Float16* hWpq = alloc(DD); _Float16* hWpk = alloc(DD); _Float16* hWpv = alloc(DD);
    _Float16* hWd  = alloc(DD);
    _Float16* hWm1 = alloc((size_t)D * DFF);
    _Float16* hWm2 = alloc((size_t)DFF * D);
    _Float16* hWml = alloc(DD);
    _Float16* dup  = alloc(BSD);
    _Float16* loc2 = alloc(BSD);
    _Float16* glb  = alloc(BSD);
    _Float16* qg   = alloc(BSD);
    _Float16* kg   = alloc(BSD);
    _Float16* vg   = alloc(BSD);
    _Float16* qp   = alloc(BSD);
    _Float16* kp   = alloc(BSD);
    _Float16* vp   = alloc(BSD);
    _Float16* Gp   = alloc(BHSS);
    _Float16* Pp   = alloc(BHSS);
    _Float16* E    = alloc(BHSS);
    _Float16* ctx  = alloc(BSD);
    _Float16* x1   = alloc(BSD);
    _Float16* h1   = alloc((size_t)B_ * S * DFF);
    _Float16* x2   = alloc(BSD);

    auto conv = [&](const float* src, _Float16* dst, size_t n) {
        int blocks = (int)((n + 255) / 256);
        if (blocks > 8192) blocks = 8192;
        f32_to_f16_kernel<<<blocks, 256, 0, stream>>>(src, dst, n);
    };

    conv(gf, hgf, BSD); conv(lf, hlf, BSD); conv(tf, htf, BSD);
    conv(Wl1, hWl1, DD); conv(Wl2, hWl2, DD); conv(Wg, hWg, DD);
    conv(Wgq, hWgq, DD); conv(Wgk, hWgk, DD); conv(Wgv, hWgv, DD);
    conv(Wpq, hWpq, DD); conv(Wpk, hWpk, DD); conv(Wpv, hWpv, DD);
    conv(Wd, hWd, DD);
    conv(Wm1, hWm1, (size_t)D * DFF);
    conv(Wm2, hWm2, (size_t)DFF * D);
    conv(Wml, hWml, DD);

    const int M = B_ * S;          // 16384 token rows
    const float one = 1.0f;
    const float scl = 0.125f;      // 1/sqrt(DH)

    // ---- embedding + QKV projections (bias epilogue, f16 out) ----
    launch_gemm<1,128,128>(hlf,  hWl1, bl1, dup,  M, D, D, D, D, D, 0,0,0,0,0,0, 1, one, 1, stream);
    launch_gemm<1,128,128>(hlf,  hWl2, bl2, loc2, M, D, D, D, D, D, 0,0,0,0,0,0, 1, one, 1, stream);
    launch_gemm<1,128,128>(hgf,  hWg,  bg,  glb,  M, D, D, D, D, D, 0,0,0,0,0,0, 1, one, 1, stream);
    launch_gemm<1,128,128>(glb,  hWgq, bgq, qg,   M, D, D, D, D, D, 0,0,0,0,0,0, 1, one, 1, stream);
    launch_gemm<1,128,128>(loc2, hWgk, bgk, kg,   M, D, D, D, D, D, 0,0,0,0,0,0, 1, one, 1, stream);
    launch_gemm<1,128,128>(loc2, hWgv, bgv, vg,   M, D, D, D, D, D, 0,0,0,0,0,0, 1, one, 1, stream);
    launch_gemm<1,128,128>(dup,  hWpq, bpq, qp,   M, D, D, D, D, D, 0,0,0,0,0,0, 1, one, 1, stream);
    launch_gemm<1,128,128>(htf,  hWpk, bpk, kp,   M, D, D, D, D, D, 0,0,0,0,0,0, 1, one, 1, stream);
    launch_gemm<1,128,128>(htf,  hWpv, bpv, vp,   M, D, D, D, D, D, 0,0,0,0,0,0, 1, one, 1, stream);

    // ---- scores: Q @ K^T * scale   (batched over B*H, head = 64-col slice) ----
    const long sdB = (long)S * D;
    const long sdH = (long)DH;
    const long psB = (long)H * SS;
    const long psH = (long)SS;
    launch_gemm<16,128,128>(qg, kg, nullptr, Gp, S, S, DH, D, D, S,
                            sdB, sdH, sdB, sdH, psB, psH, H, scl, B_ * H, stream);
    launch_gemm<16,128,128>(qp, kp, nullptr, Pp, S, S, DH, D, D, S,
                            sdB, sdH, sdB, sdH, psB, psH, H, scl, B_ * H, stream);

    // ---- softmax (in place, one block per row) ----
    softmax_rows_kernel<<<B_ * H * S, 256, 0, stream>>>(Gp, S);
    softmax_rows_kernel<<<B_ * H * S, 256, 0, stream>>>(Pp, S);

    // ---- bridging matmul: E = Gp @ Pp  (the 550-GFLOP hot spot) ----
    launch_gemm<0,128,128>(Gp, Pp, nullptr, E, S, S, S, S, S, S,
                           psB, psH, psB, psH, psB, psH, H, one, B_ * H, stream);

    // ---- contexts: ctx = Gp @ vg ; ctx += E @ vp ----
    launch_gemm<0,128,64>(Gp, vg, nullptr, ctx, S, DH, S, S, D, D,
                          psB, psH, sdB, sdH, sdB, sdH, H, one, B_ * H, stream);
    launch_gemm<4,128,64>(E, vp, nullptr, ctx, S, DH, S, S, D, D,
                          psB, psH, sdB, sdH, sdB, sdH, H, one, B_ * H, stream);

    // ---- output projection + MLP + final linear ----
    launch_gemm<1,128,128>(ctx, hWd,  bd,  x1,    M, D,   D,   D,   D,   D,   0,0,0,0,0,0, 1, one, 1, stream);
    launch_gemm<3,128,128>(x1,  hWm1, bm1, h1,    M, DFF, D,   D,   DFF, DFF, 0,0,0,0,0,0, 1, one, 1, stream);
    launch_gemm<1,128,128>(h1,  hWm2, bm2, x2,    M, D,   DFF, DFF, D,   D,   0,0,0,0,0,0, 1, one, 1, stream);
    launch_gemm<9,128,128>(x2,  hWml, bml, d_out, M, D,   D,   D,   D,   D,   0,0,0,0,0,0, 1, one, 1, stream);
}
